// Attention_66348654789030
// MI455X (gfx1250) — compile-verified, matrix operand order checked by reading, not run on hardware
//
#include <hip/hip_runtime.h>
#include <hip/hip_bf16.h>

// ---------------------------------------------------------------------------
// Problem constants (B=8, N=1024, C=1024, H=16, D=64)
// ---------------------------------------------------------------------------
#define BATCH 8
#define SEQ   1024
#define CDIM  1024
#define HEADS 16
#define HDIM  64
#define BH    (BATCH * HEADS)          // 128
#define MROWS (BATCH * SEQ)            // 8192

typedef __attribute__((ext_vector_type(16))) __bf16 v16bf;
typedef __attribute__((ext_vector_type(8)))  float  v8f;

// float -> bf16 (round to nearest even), stored as ushort
__device__ __forceinline__ unsigned short f2bf(float f) {
    unsigned int u = __float_as_uint(f);
    u += 0x7FFFu + ((u >> 16) & 1u);
    return (unsigned short)(u >> 16);
}

// Wave-local LDS fence: DS ops are in-order per wave; wait for completion.
__device__ __forceinline__ void lds_fence() {
    asm volatile("s_wait_dscnt 0" ::: "memory");
}

// 16x32 bf16 fragment (A or B operand of V_WMMA_F32_16X16X32_BF16)
// Lanes 0-15 hold K[0..7]/K[16..23], lanes 16-31 K[8..15]/K[24..31]
// for row M = lane%16 -> two contiguous 16B chunks per lane.
union Frag {
    v16bf v;
    uint4 q[2];
};

// Load a fragment from a row-major [*, ld] bf16 matrix: rows row0..row0+15,
// K-slice k0..k0+31. All addresses 16B aligned (ld multiples of 8, k0 of 32).
__device__ __forceinline__ Frag load_frag(const unsigned short* __restrict__ src,
                                          int ld, int row0, int k0, int lane) {
    const int l  = lane & 15;
    const int hi = lane >> 4;
    const unsigned short* p = src + (size_t)(row0 + l) * ld + k0 + hi * 8;
    Frag f;
    f.q[0] = *(const uint4*)(p);
    f.q[1] = *(const uint4*)(p + 16);
    return f;
}

__device__ __forceinline__ v8f wmma_bf16(const Frag& a, const Frag& b, v8f c) {
    return __builtin_amdgcn_wmma_f32_16x16x32_bf16(
        /*neg_a=*/false, a.v, /*neg_b=*/false, b.v,
        /*c_mod=*/(short)0, c, /*reuse_a=*/false, /*reuse_b=*/false);
}

// ---------------------------------------------------------------------------
// Async staging of a 64-row x 32-col bf16 weight tile (4096 B) into LDS.
// 128 threads x 2 chunks x 16 B, via CDNA5 GLOBAL_LOAD_ASYNC_TO_LDS_B128
// (GV mode: 64-bit global addr in VGPR pair, LDS byte offset in VGPR).
// Each wave issues exactly 2 async instructions -> ASYNCcnt += 2.
// ---------------------------------------------------------------------------
__device__ __forceinline__ void stage_async(const unsigned short* __restrict__ W,
                                            int ldw, int n0, int k0,
                                            unsigned short* ldsDst, int tid) {
#pragma unroll
    for (int pass = 0; pass < 2; ++pass) {
        const int c    = tid + pass * 128;    // chunk id 0..255
        const int row  = c >> 2;              // 0..63
        const int quad = c & 3;               // 16B quarter of the 64B row
        const unsigned long long g =
            (unsigned long long)(size_t)(W + (size_t)(n0 + row) * ldw + k0 + quad * 8);
        const unsigned int l =
            (unsigned int)(size_t)(ldsDst + row * 32 + quad * 8);
        asm volatile("global_load_async_to_lds_b128 %0, %1, off"
                     :: "v"(l), "v"(g) : "memory");
    }
}

// Block-cooperative GEMM mainloop: block = 4 waves; each wave owns 16 rows,
// all share a double-buffered 64x32 LDS weight tile. acc[t] = 16x16 tile at
// columns n0 + t*16.
__device__ __forceinline__ void gemm_tile_lds(const unsigned short* __restrict__ A,
                                              const unsigned short* __restrict__ W,
                                              unsigned short* ldsB,   // [2][64*32]
                                              int m0, int n0, int K, int lda,
                                              v8f acc[4], int tid, int lane) {
    stage_async(W, lda, n0, 0, ldsB, tid);
    for (int k0 = 0; k0 < K; k0 += 32) {
        const int  buf  = (k0 >> 5) & 1;
        const bool more = (k0 + 32) < K;
        if (more) {
            stage_async(W, lda, n0, k0 + 32, ldsB + (buf ^ 1) * (64 * 32), tid);
            asm volatile("s_wait_asynccnt 0x2" ::: "memory");   // current tile done
        } else {
            asm volatile("s_wait_asynccnt 0x0" ::: "memory");
        }
        __syncthreads();                                        // cross-wave visibility

        Frag a = load_frag(A, lda, m0, k0, lane);
        const unsigned short* bsrc = ldsB + buf * (64 * 32);
#pragma unroll
        for (int t = 0; t < 4; ++t) {
            Frag b = load_frag(bsrc, 32, t * 16, 0, lane);      // ds_load_b128 x2
            acc[t] = wmma_bf16(a, b, acc[t]);
        }
        __syncthreads();    // all waves done reading before next overwrite
    }
}

// ---------------------------------------------------------------------------
// fp32 -> bf16 conversion (grid-stride)
// ---------------------------------------------------------------------------
__global__ void cvt_bf16_kernel(const float* __restrict__ src,
                                unsigned short* __restrict__ dst, size_t n) {
    size_t i = (size_t)blockIdx.x * blockDim.x + threadIdx.x;
    size_t stride = (size_t)gridDim.x * blockDim.x;
    for (; i < n; i += stride) dst[i] = f2bf(src[i]);
}

// ---------------------------------------------------------------------------
// QKV GEMM:  Y[m,n] = sum_k x[m,k] * qkv_w[n,k]   (M=8192, N=3072, K=1024)
// Block: 64 rows x 64 cols. Epilogue scatters into Q/K (BH,N,D) and
// transposed Vt (BH,D,N), all bf16.
// ---------------------------------------------------------------------------
__global__ void gemm_qkv_kernel(const unsigned short* __restrict__ A,
                                const unsigned short* __restrict__ W,
                                unsigned short* __restrict__ Qp,
                                unsigned short* __restrict__ Kp,
                                unsigned short* __restrict__ Vtp) {
    __shared__ unsigned short ldsB[2][64 * 32];     // 8 KB double buffer
    const int tid  = threadIdx.x;
    const int lane = tid & 31;
    const int wave = tid >> 5;
    const int tilesN = (3 * CDIM) / 64;             // 48
    const int bm = blockIdx.x / tilesN;
    const int bn = blockIdx.x % tilesN;
    const int m0 = bm * 64 + wave * 16;
    const int n0 = bn * 64;

    v8f acc[4] = {};
    gemm_tile_lds(A, W, &ldsB[0][0], m0, n0, CDIM, CDIM, acc, tid, lane);

    const int l  = lane & 15;
    const int hi = lane >> 4;
#pragma unroll
    for (int t = 0; t < 4; ++t) {
#pragma unroll
        for (int i = 0; i < 8; ++i) {
            const int n = n0 + t * 16 + l;
            const int m = m0 + i + hi * 8;
            const unsigned short bv = f2bf(acc[t][i]);
            const int which = n >> 10;              // 0=Q 1=K 2=V
            const int c  = n & (CDIM - 1);
            const int h  = c >> 6;
            const int d  = c & (HDIM - 1);
            const int b  = m >> 10;
            const int nn = m & (SEQ - 1);
            const size_t bh = (size_t)(b * HEADS + h);
            if (which == 0)
                Qp[bh * (SEQ * HDIM) + (size_t)nn * HDIM + d] = bv;
            else if (which == 1)
                Kp[bh * (SEQ * HDIM) + (size_t)nn * HDIM + d] = bv;
            else
                Vtp[bh * (SEQ * HDIM) + (size_t)d * SEQ + nn] = bv;
        }
    }
}

// ---------------------------------------------------------------------------
// Flash attention: one wave per (b,h, 16-query-row block). Streams keys in
// blocks of 32 with online softmax via per-wave LDS.
// ---------------------------------------------------------------------------
__global__ void attn_kernel(const unsigned short* __restrict__ Q,
                            const unsigned short* __restrict__ K,
                            const unsigned short* __restrict__ Vt,
                            unsigned short* __restrict__ O) {
    __shared__ float          sS[4][16][32];
    __shared__ unsigned short sP[4][16][32];
    __shared__ float          sM[4][16];
    __shared__ float          sL[4][16];
    __shared__ float          sA[4][16];
    __shared__ float          sRed[4][2][16];

    const int w    = threadIdx.x >> 5;
    const int lane = threadIdx.x & 31;
    const int l    = lane & 15;
    const int hi   = lane >> 4;
    const int gw   = blockIdx.x * 4 + w;
    const int bh = gw >> 6;                 // SEQ/16 = 64 row blocks
    const int m0 = (gw & 63) * 16;

    const unsigned short* Qb = Q  + (size_t)bh * SEQ * HDIM;
    const unsigned short* Kb = K  + (size_t)bh * SEQ * HDIM;
    const unsigned short* Vb = Vt + (size_t)bh * HDIM * SEQ;

    // Q fragments are reused across the whole key loop
    const Frag aQ0 = load_frag(Qb, HDIM, m0, 0,  lane);
    const Frag aQ1 = load_frag(Qb, HDIM, m0, 32, lane);

    v8f o[4] = {};
    if (hi == 0) { sM[w][l] = -__builtin_inff(); sL[w][l] = 0.0f; }
    lds_fence();

    const float scale = 0.125f;             // 1/sqrt(64)

    for (int kb = 0; kb < SEQ; kb += 32) {
        // ---- S = Q @ K^T for 32 keys (two 16x16 tiles) ----
        v8f s0 = {}, s1 = {};
        {
            Frag b0 = load_frag(Kb, HDIM, kb,      0,  lane); s0 = wmma_bf16(aQ0, b0, s0);
            Frag b1 = load_frag(Kb, HDIM, kb,      32, lane); s0 = wmma_bf16(aQ1, b1, s0);
            Frag b2 = load_frag(Kb, HDIM, kb + 16, 0,  lane); s1 = wmma_bf16(aQ0, b2, s1);
            Frag b3 = load_frag(Kb, HDIM, kb + 16, 32, lane); s1 = wmma_bf16(aQ1, b3, s1);
        }
        // ---- stash scaled scores in LDS ----
#pragma unroll
        for (int i = 0; i < 8; ++i) {
            const int r = i + hi * 8;
            sS[w][r][l]      = s0[i] * scale;
            sS[w][r][l + 16] = s1[i] * scale;
        }
        lds_fence();

        // ---- row max (2 lanes per row, 16 cols each) ----
        float pm = -__builtin_inff();
#pragma unroll
        for (int c = 0; c < 16; ++c) pm = fmaxf(pm, sS[w][l][hi * 16 + c]);
        sRed[w][hi][l] = pm;
        lds_fence();
        if (hi == 0) {
            const float mrow = fmaxf(sRed[w][0][l], sRed[w][1][l]);
            const float mold = sM[w][l];
            const float mnew = fmaxf(mold, mrow);
            sA[w][l] = __expf(mold - mnew);
            sM[w][l] = mnew;
        }
        lds_fence();

        // ---- P = exp(S - m), partial row sums, bf16 P into LDS ----
        const float mnew = sM[w][l];
        float ps = 0.0f;
#pragma unroll
        for (int c = 0; c < 16; ++c) {
            const int cc = hi * 16 + c;
            const float p = __expf(sS[w][l][cc] - mnew);
            sP[w][l][cc] = f2bf(p);
            ps += p;
        }
        sRed[w][hi][l] = ps;
        lds_fence();
        if (hi == 0)
            sL[w][l] = sL[w][l] * sA[w][l] + sRed[w][0][l] + sRed[w][1][l];

        // ---- rescale running output by alpha (per-row broadcast) ----
#pragma unroll
        for (int i = 0; i < 8; ++i) {
            const float al = sA[w][i + hi * 8];
#pragma unroll
            for (int t = 0; t < 4; ++t) o[t][i] *= al;
        }

        // ---- O += P @ V  (P from LDS, V columns from transposed Vt) ----
        Frag aP = load_frag(&sP[w][0][0], 32, 0, 0, lane);
#pragma unroll
        for (int t = 0; t < 4; ++t) {
            Frag bV = load_frag(Vb, SEQ, t * 16, kb, lane);
            o[t] = wmma_bf16(aP, bV, o[t]);
        }
    }
    lds_fence();

    // ---- finalize: divide by l, scatter into [B,N,C] bf16 (col h*64) ----
    const int b = bh >> 4;
    const int h = bh & 15;
#pragma unroll
    for (int i = 0; i < 8; ++i) {
        const int r = i + hi * 8;
        const float linv = 1.0f / sL[w][r];
        const size_t rowbase = ((size_t)(b * SEQ + m0 + r)) * CDIM + h * HDIM;
#pragma unroll
        for (int t = 0; t < 4; ++t)
            O[rowbase + t * 16 + l] = f2bf(o[t][i] * linv);
    }
}

// ---------------------------------------------------------------------------
// Proj GEMM: out[m,n] = sum_k attn[m,k] * proj_w[n,k] + bias[n]   (fp32 out)
// ---------------------------------------------------------------------------
__global__ void gemm_proj_kernel(const unsigned short* __restrict__ A,
                                 const unsigned short* __restrict__ W,
                                 const float* __restrict__ bias,
                                 float* __restrict__ out) {
    __shared__ unsigned short ldsB[2][64 * 32];     // 8 KB double buffer
    const int tid  = threadIdx.x;
    const int lane = tid & 31;
    const int wave = tid >> 5;
    const int tilesN = CDIM / 64;                   // 16
    const int bm = blockIdx.x / tilesN;
    const int bn = blockIdx.x % tilesN;
    const int m0 = bm * 64 + wave * 16;
    const int n0 = bn * 64;

    v8f acc[4] = {};
    gemm_tile_lds(A, W, &ldsB[0][0], m0, n0, CDIM, CDIM, acc, tid, lane);

    const int l  = lane & 15;
    const int hi = lane >> 4;
#pragma unroll
    for (int t = 0; t < 4; ++t) {
        const int n = n0 + t * 16 + l;
        const float bv = bias[n];
#pragma unroll
        for (int i = 0; i < 8; ++i) {
            const int m = m0 + i + hi * 8;
            out[(size_t)m * CDIM + n] = acc[t][i] + bv;
        }
    }
}

// ---------------------------------------------------------------------------
// Launch
// ---------------------------------------------------------------------------
extern "C" void kernel_launch(void* const* d_in, const int* in_sizes, int n_in,
                              void* d_out, int out_size, void* d_ws, size_t ws_size,
                              hipStream_t stream) {
    const float* x      = (const float*)d_in[0];   // [8,1024,1024]
    const float* qkv_w  = (const float*)d_in[1];   // [3072,1024]
    const float* proj_w = (const float*)d_in[2];   // [1024,1024]
    const float* proj_b = (const float*)d_in[3];   // [1024]
    float* out = (float*)d_out;

    // workspace carve-up (bf16/ushort elements)
    unsigned short* ws    = (unsigned short*)d_ws;
    unsigned short* xb    = ws;                                  // 8192*1024
    unsigned short* wqkv  = xb    + (size_t)MROWS * CDIM;        // 3072*1024
    unsigned short* wproj = wqkv  + (size_t)3 * CDIM * CDIM;     // 1024*1024
    unsigned short* Qb    = wproj + (size_t)CDIM * CDIM;         // BH*N*D
    unsigned short* Kb    = Qb    + (size_t)BH * SEQ * HDIM;
    unsigned short* Vt    = Kb    + (size_t)BH * SEQ * HDIM;
    unsigned short* attnO = Vt    + (size_t)BH * SEQ * HDIM;     // 8192*1024

    cvt_bf16_kernel<<<4096, 256, 0, stream>>>(x,      xb,    (size_t)MROWS * CDIM);
    cvt_bf16_kernel<<<2048, 256, 0, stream>>>(qkv_w,  wqkv,  (size_t)3 * CDIM * CDIM);
    cvt_bf16_kernel<<<1024, 256, 0, stream>>>(proj_w, wproj, (size_t)CDIM * CDIM);

    // QKV GEMM: (8192/64)*(3072/64) = 6144 blocks of 128 threads
    gemm_qkv_kernel<<<6144, 128, 0, stream>>>(xb, wqkv, Qb, Kb, Vt);

    // Attention: 128 heads * 64 row-blocks = 8192 waves -> 2048 blocks
    attn_kernel<<<2048, 128, 0, stream>>>(Qb, Kb, Vt, attnO);

    // Proj GEMM: (8192/64)*(1024/64) = 2048 blocks
    gemm_proj_kernel<<<2048, 128, 0, stream>>>(attnO, wproj, proj_b, out);
}
